// CAMixer_84129819394219
// MI455X (gfx1250) — compile-verified, matrix operand order checked by reading, not run on hardware
//
#include <hip/hip_runtime.h>
#include <cmath>

// ---------------------------------------------------------------------------
// CAMixer forward for MI455X (gfx1250, wave32, WMMA).
// B=4, C=60, H=W=256, WS=8 -> 4096 windows of 64 tokens x 60 channels.
// All matmuls via v_wmma_f32_16x16x32_f16 (C padded 60->64).
// ---------------------------------------------------------------------------

#define BB    4
#define CC    60
#define HH    256
#define WW2   256
#define HWH   65536            // H*W
#define NWIN  1024             // (256/8)^2
#define LDH   72               // LDS row stride in halves; 144B = 9*16 -> 16B aligned rows

typedef __attribute__((ext_vector_type(16))) _Float16 v16h;
typedef __attribute__((ext_vector_type(8)))  _Float16 h8;
typedef __attribute__((ext_vector_type(8)))  float    v8f;

// Load a 16x32 f16 WMMA A/B fragment from LDS laid out row-major [row][LDH].
// ISA 16-bit A layout (CDNA5 ISA): lanes 0-15 hold row M=lane, K = k0..k0+7
// (VGPR0-3) and K = k0+16..k0+23 (VGPR4-7); lanes 16-31 hold K+8 / K+24.
// Each half-fragment is one aligned 16-byte LDS read (ds_load_b128).
__device__ __forceinline__ v16h ld_frag(const _Float16* base, int row0, int k0) {
  const int lane = threadIdx.x & 31;
  const _Float16* p = base + (row0 + (lane & 15)) * LDH + k0 + ((lane >> 4) << 3);
  h8 lo = *(const h8*)p;          // K = kk .. kk+7
  h8 hi = *(const h8*)(p + 16);   // K = kk+16 .. kk+23
  v16h f;
#pragma unroll
  for (int i = 0; i < 8; ++i) { f[i] = lo[i]; f[i + 8] = hi[i]; }
  return f;
}

// ---------------------------------------------------------------------------
// Convert a 60x60 f32 weight matrix to 64x64 f16 (zero padded), row-major.
// ---------------------------------------------------------------------------
__global__ void __launch_bounds__(256) k_cvt_w(const float* __restrict__ src,
                                               _Float16* __restrict__ dst) {
  const int i = blockIdx.x * 256 + threadIdx.x;   // 0..4095
  const int o = i >> 6, c = i & 63;
  dst[i] = (_Float16)((o < CC && c < CC) ? src[o * CC + c] : 0.0f);
}

// ---------------------------------------------------------------------------
// 1x1 conv as GEMM: Y[b,o,hw] = sum_c X[b,c,hw] * W[o,c] (+ bias[o]).
// 256 threads (8 waves), 128 pixels/block. Per wave: 16x64 C tile = 8 WMMAs.
// ---------------------------------------------------------------------------
__global__ void __launch_bounds__(256) k_gemm_c1x1(const float* __restrict__ X,
                                                   const _Float16* __restrict__ Wh,
                                                   const float* __restrict__ bias,
                                                   float* __restrict__ Y,
                                                   int use_bias) {
  __shared__ __align__(16) _Float16 sX[128 * LDH];
  __shared__ __align__(16) _Float16 sW[64 * LDH];
  const int t = threadIdx.x;
  const int pix0 = blockIdx.x * 128;

  __builtin_prefetch(Wh, 0, 3);                 // global_prefetch_b8 of weight panel

  for (int i = t; i < 64 * 64; i += 256)
    sW[(i >> 6) * LDH + (i & 63)] = Wh[i];

  for (int i = t; i < 128 * 64; i += 256) {
    const int pl = i >> 6, c = i & 63;
    float xv = 0.0f;
    if (c < CC) {
      const int pg = pix0 + pl;
      const int b = pg >> 16, hw = pg & (HWH - 1);
      xv = X[((b * CC + c) << 16) + hw];
    }
    sX[pl * LDH + c] = (_Float16)xv;
  }
  __syncthreads();

  const int wave = t >> 5, lane = t & 31;
  const int row0 = wave * 16;
  const int n_lane = lane & 15;
  const int m_base = (lane >> 4) << 3;          // C-frag: lanes16-31 hold M+8

#pragma unroll
  for (int nt = 0; nt < 4; ++nt) {
    v8f acc = {};
#pragma unroll
    for (int kt = 0; kt < 2; ++kt) {
      v16h a  = ld_frag(sX, row0,    kt * 32);
      v16h bm = ld_frag(sW, nt * 16, kt * 32);
      acc = __builtin_amdgcn_wmma_f32_16x16x32_f16(false, a, false, bm,
                                                   (short)0, acc, false, false);
    }
    const int o = nt * 16 + n_lane;
    if (o < CC) {
      const float bv = use_bias ? bias[o] : 0.0f;
#pragma unroll
      for (int r = 0; r < 8; ++r) {
        const int pg = pix0 + row0 + m_base + r;
        const int b = pg >> 16, hw = pg & (HWH - 1);
        Y[((b * CC + o) << 16) + hw] = acc[r] + bv;
      }
    }
  }
}

// ---------------------------------------------------------------------------
// PredictorLG: per-window routing mask + per-window partial sums for ca.
// One block (64 threads = 1 pixel/thread) per 8x8 window.
// ---------------------------------------------------------------------------
__device__ __forceinline__ float wang(unsigned s) {
  s ^= s >> 16; s *= 0x7feb352dU; s ^= s >> 15; s *= 0x846ca68bU; s ^= s >> 16;
  return (float)(s >> 8) * (1.0f / 16777216.0f) + 1e-10f;
}

__global__ void __launch_bounds__(64) k_predictor(
    const float* __restrict__ V,  const float* __restrict__ CG,
    const float* __restrict__ rin_w, const float* __restrict__ rin_b,
    const float* __restrict__ rln_g, const float* __restrict__ rln_b,
    const float* __restrict__ rm1_w, const float* __restrict__ rm1_b,
    const float* __restrict__ rm2_w, const float* __restrict__ rm2_b,
    float* __restrict__ maskb, float* __restrict__ psum) {
  __shared__ float sc[64][68];       // cond per pixel (60 v + 2 cg + 2 wind), padded
  __shared__ float srw[16 * 64];     // rin_w
  __shared__ float sred[16][64];     // activated t for ca reduction
  __shared__ float stm[64];          // per-pixel channel-mean
  __shared__ float sh1[8];
  const int t = threadIdx.x;
  const int win = blockIdx.x;
  const int b = win >> 10, wi = win & 1023;
  const int wy = wi >> 5, wx = wi & 31;
  const int ty = t >> 3, tx = t & 7;
  const int hw = ((wy * 8 + ty) << 8) + wx * 8 + tx;

  for (int i = t; i < 16 * 64; i += 64) srw[i] = rin_w[i];
  for (int c = 0; c < CC; ++c) sc[t][c] = V[((b * CC + c) << 16) + hw];
  sc[t][60] = CG[((b * 2 + 0) << 16) + hw];
  sc[t][61] = CG[((b * 2 + 1) << 16) + hw];
  sc[t][62] = -1.0f + (2.0f / 7.0f) * (float)ty;   // gy (ij-indexed meshgrid)
  sc[t][63] = -1.0f + (2.0f / 7.0f) * (float)tx;   // gx
  __syncthreads();

  float tj[16];
  float mu = 0.f, m2 = 0.f;
#pragma unroll
  for (int j = 0; j < 16; ++j) {
    float a = rin_b[j];
    for (int c = 0; c < 64; ++c) a += srw[j * 64 + c] * sc[t][c];
    tj[j] = a; mu += a; m2 += a * a;
  }
  mu *= (1.f / 16.f);
  m2 = m2 * (1.f / 16.f) - mu * mu;                // population variance
  const float inv = rsqrtf(m2 + 1e-5f);
  float tm = 0.f;
#pragma unroll
  for (int j = 0; j < 16; ++j) {
    float v = (tj[j] - mu) * inv * rln_g[j] + rln_b[j];
    v = (v >= 0.f) ? v : 0.1f * v;                 // leaky 0.1
    sred[j][t] = v;
    tm += v;
  }
  stm[t] = tm * (1.f / 16.f);
  __syncthreads();

  if (t < 16) {                                    // deterministic partial sums
    float s = 0.f;
    for (int p = 0; p < 64; ++p) s += sred[t][p];
    psum[win * 16 + t] = s;
  }
  if (t < 8) {
    float a = rm1_b[t];
    for (int e = 0; e < 64; ++e) a += rm1_w[t * 64 + e] * stm[e];
    sh1[t] = (a >= 0.f) ? a : 0.1f * a;
  }
  __syncthreads();

  if (t == 0) {
    float l0 = rm2_b[0], l1 = rm2_b[1];
    for (int o = 0; o < 8; ++o) { l0 += rm2_w[o] * sh1[o]; l1 += rm2_w[8 + o] * sh1[o]; }
    const float mx = fmaxf(l0, l1);
    const float e0 = expf(l0 - mx), e1 = expf(l1 - mx);
    const float p0 = e0 / (e0 + e1), p1 = e1 / (e0 + e1);
    const float g0 = -logf(-logf(wang((unsigned)win * 2u + 1u)));
    const float g1 = -logf(-logf(wang((unsigned)win * 2u + 2u)));
    // hard gumbel-softmax straight-through -> forward value is one-hot channel 0
    maskb[win] = (p0 + g0 >= p1 + g1) ? 1.0f : 0.0f;
  }
}

// ---------------------------------------------------------------------------
// Channel attention: ca[b,o] = sigmoid(rca_b + rca_w . mean_{h,w} t)
// ---------------------------------------------------------------------------
__global__ void __launch_bounds__(64) k_ca(const float* __restrict__ psum,
                                           const float* __restrict__ rca_w,
                                           const float* __restrict__ rca_b,
                                           float* __restrict__ cab) {
  __shared__ float tj[16];
  const int b = blockIdx.x, t = threadIdx.x;
  if (t < 16) {
    float s = 0.f;
    for (int wn = 0; wn < NWIN; ++wn) s += psum[(b * NWIN + wn) * 16 + t];
    tj[t] = s * (1.0f / (float)HWH);
  }
  __syncthreads();
  if (t < CC) {
    float a = rca_b[t];
    for (int j = 0; j < 16; ++j) a += rca_w[t * 16 + j] * tj[j];
    cab[b * CC + t] = 1.0f / (1.0f + expf(-a));
  }
}

// ---------------------------------------------------------------------------
// 3x3 group conv, groups=6 (10 in-channels per out-channel): vs = gconv(v)+b
// ---------------------------------------------------------------------------
__global__ void __launch_bounds__(256) k_gconv(const float* __restrict__ V,
                                               const float* __restrict__ ls_w,
                                               const float* __restrict__ ls_b,
                                               float* __restrict__ VS) {
  const int idx = blockIdx.x * 256 + threadIdx.x;   // == ((b*CC+c)<<16)+hw
  const int hw = idx & (HWH - 1);
  const int bc = idx >> 16;
  const int c = bc % CC, b = bc / CC;
  const int h = hw >> 8, w = hw & 255;
  const int g = c / 10;
  float acc = ls_b[c];
  for (int i = 0; i < 10; ++i) {
    const float* vp = V + ((b * CC + g * 10 + i) << 16);
    const float* wp = ls_w + (c * 10 + i) * 9;
#pragma unroll
    for (int ky = 0; ky < 3; ++ky) {
      const int hh = h + ky - 1;
      if ((unsigned)hh >= HH) continue;
#pragma unroll
      for (int kx = 0; kx < 3; ++kx) {
        const int ww = w + kx - 1;
        if ((unsigned)ww >= WW2) continue;
        acc += vp[(hh << 8) + ww] * wp[ky * 3 + kx];
      }
    }
  }
  VS[idx] = acc;
}

// ---------------------------------------------------------------------------
// Window attention, one block per window (128 threads = 4 waves).
// q1 = mk*(x.Wq)+bq ; k1 = mk*(x.Wk)+bk ; v1 = mk*v   (linearity of conv1x1)
// S = q1.k1^T (WMMA) -> softmax (LDS) -> P.v1 (WMMA) ; out += (1-mk)*vs
// ---------------------------------------------------------------------------
__global__ void __launch_bounds__(128) k_attn(const float* __restrict__ XQ,
                                              const float* __restrict__ XK,
                                              const float* __restrict__ V,
                                              const float* __restrict__ VS,
                                              const float* __restrict__ maskb,
                                              const float* __restrict__ pq_b,
                                              const float* __restrict__ pk_b,
                                              float* __restrict__ AO) {
  __shared__ __align__(16) _Float16 sQ[64 * LDH];
  __shared__ __align__(16) _Float16 sK[64 * LDH];
  __shared__ __align__(16) _Float16 sVT[64 * LDH];   // v transposed: [channel][token]
  __shared__ __align__(16) _Float16 sP[64 * LDH];
  __shared__ float sS[64 * 65];
  const int t = threadIdx.x;
  const int win = blockIdx.x;
  const int b = win >> 10, wi = win & 1023;
  const int wy = wi >> 5, wx = wi & 31;
  const float mk = maskb[win];

  for (int i = t; i < 64 * 64; i += 128) {
    const int tok = i >> 6, c = i & 63;
    const int hw = ((wy * 8 + (tok >> 3)) << 8) + wx * 8 + (tok & 7);
    _Float16 qv = (_Float16)0.f, kv = (_Float16)0.f, vv = (_Float16)0.f;
    if (c < CC) {
      const int gidx = ((b * CC + c) << 16) + hw;
      qv = (_Float16)(mk * XQ[gidx] + pq_b[c]);
      kv = (_Float16)(mk * XK[gidx] + pk_b[c]);
      vv = (_Float16)(mk * V[gidx]);
    }
    sQ[tok * LDH + c] = qv;
    sK[tok * LDH + c] = kv;
    sVT[c * LDH + tok] = vv;
  }
  __syncthreads();

  const int wave = t >> 5, lane = t & 31;
  const int row0 = wave * 16;
  const int n_lane = lane & 15;
  const int m_base = (lane >> 4) << 3;

  // S = Q . K^T   (K-dim = channels, padded to 64)
#pragma unroll
  for (int nt = 0; nt < 4; ++nt) {
    v8f acc = {};
#pragma unroll
    for (int kt = 0; kt < 2; ++kt) {
      v16h a  = ld_frag(sQ, row0,    kt * 32);
      v16h bm = ld_frag(sK, nt * 16, kt * 32);
      acc = __builtin_amdgcn_wmma_f32_16x16x32_f16(false, a, false, bm,
                                                   (short)0, acc, false, false);
    }
#pragma unroll
    for (int r = 0; r < 8; ++r)
      sS[(row0 + m_base + r) * 65 + nt * 16 + n_lane] = acc[r];
  }
  __syncthreads();

  // row softmax (two-pass to keep VGPR pressure low)
  if (t < 64) {
    float mx = -1e30f;
    for (int j = 0; j < 64; ++j) mx = fmaxf(mx, sS[t * 65 + j]);
    float sum = 0.f;
    for (int j = 0; j < 64; ++j) sum += expf(sS[t * 65 + j] - mx);
    const float is = 1.0f / sum;
    for (int j = 0; j < 64; ++j)
      sP[t * LDH + j] = (_Float16)(expf(sS[t * 65 + j] - mx) * is);
  }
  __syncthreads();

  // F = P . V   (K-dim = 64 tokens) ; fuse LSConv residual (1-mk)*vs
#pragma unroll
  for (int nt = 0; nt < 4; ++nt) {
    v8f acc = {};
#pragma unroll
    for (int kt = 0; kt < 2; ++kt) {
      v16h a  = ld_frag(sP,  row0,    kt * 32);
      v16h bm = ld_frag(sVT, nt * 16, kt * 32);
      acc = __builtin_amdgcn_wmma_f32_16x16x32_f16(false, a, false, bm,
                                                   (short)0, acc, false, false);
    }
    const int o = nt * 16 + n_lane;
    if (o < CC) {
#pragma unroll
      for (int r = 0; r < 8; ++r) {
        const int tok = row0 + m_base + r;
        const int hw = ((wy * 8 + (tok >> 3)) << 8) + wx * 8 + (tok & 7);
        const int gidx = ((b * CC + o) << 16) + hw;
        AO[gidx] = acc[r] + (1.0f - mk) * VS[gidx];
      }
    }
  }
}

// ---------------------------------------------------------------------------
// Depthwise 3x3 (dilation=1): cs1 = dwconv(attn_out)+b
// ---------------------------------------------------------------------------
__global__ void __launch_bounds__(256) k_dwconv1(const float* __restrict__ IN,
                                                 const float* __restrict__ Wd,
                                                 const float* __restrict__ Bd,
                                                 float* __restrict__ OUT) {
  const int idx = blockIdx.x * 256 + threadIdx.x;
  const int hw = idx & (HWH - 1);
  const int bc = idx >> 16;
  const int c = bc % CC;
  const int h = hw >> 8, w = hw & 255;
  const float* ip = IN + (bc << 16);
  const float* wp = Wd + c * 9;
  float acc = Bd[c];
#pragma unroll
  for (int ky = 0; ky < 3; ++ky) {
    const int hh = h + ky - 1;
    if ((unsigned)hh >= HH) continue;
#pragma unroll
    for (int kx = 0; kx < 3; ++kx) {
      const int ww = w + kx - 1;
      if ((unsigned)ww >= WW2) continue;
      acc += ip[(hh << 8) + ww] * wp[ky * 3 + kx];
    }
  }
  OUT[idx] = acc;
}

// ---------------------------------------------------------------------------
// Depthwise 3x3 (dilation=2) fused with exact GELU, channel-attn scale and
// residual: pre = gelu(dw2(cs1)) * ca + attn_out
// ---------------------------------------------------------------------------
__global__ void __launch_bounds__(256) k_dw2_gelu(const float* __restrict__ IN,
                                                  const float* __restrict__ Wd,
                                                  const float* __restrict__ Bd,
                                                  const float* __restrict__ AOb,
                                                  const float* __restrict__ cab,
                                                  float* __restrict__ OUT) {
  const int idx = blockIdx.x * 256 + threadIdx.x;
  const int hw = idx & (HWH - 1);
  const int bc = idx >> 16;
  const int c = bc % CC, b = bc / CC;
  const int h = hw >> 8, w = hw & 255;
  const float* ip = IN + (bc << 16);
  const float* wp = Wd + c * 9;
  float acc = Bd[c];
#pragma unroll
  for (int ky = 0; ky < 3; ++ky) {
    const int hh = h + (ky - 1) * 2;
    if ((unsigned)hh >= HH) continue;
#pragma unroll
    for (int kx = 0; kx < 3; ++kx) {
      const int ww = w + (kx - 1) * 2;
      if ((unsigned)ww >= WW2) continue;
      acc += ip[(hh << 8) + ww] * wp[ky * 3 + kx];
    }
  }
  const float g = 0.5f * acc * (1.0f + erff(acc * 0.70710678118f));  // exact gelu
  OUT[idx] = g * cab[b * CC + c] + AOb[idx];
}

// ---------------------------------------------------------------------------
// Launch
// ---------------------------------------------------------------------------
extern "C" void kernel_launch(void* const* d_in, const int* in_sizes, int n_in,
                              void* d_out, int out_size, void* d_ws, size_t ws_size,
                              hipStream_t stream) {
  (void)in_sizes; (void)n_in; (void)out_size; (void)ws_size;
  const float* x     = (const float*)d_in[0];
  const float* cg    = (const float*)d_in[1];
  const float* pv_w  = (const float*)d_in[2];  const float* pv_b  = (const float*)d_in[3];
  const float* pq_w  = (const float*)d_in[4];  const float* pq_b  = (const float*)d_in[5];
  const float* pk_w  = (const float*)d_in[6];  const float* pk_b  = (const float*)d_in[7];
  const float* cs1_w = (const float*)d_in[8];  const float* cs1_b = (const float*)d_in[9];
  const float* cs2_w = (const float*)d_in[10]; const float* cs2_b = (const float*)d_in[11];
  const float* po_w  = (const float*)d_in[12]; const float* po_b  = (const float*)d_in[13];
  const float* ls_w  = (const float*)d_in[14]; const float* ls_b  = (const float*)d_in[15];
  const float* rin_w = (const float*)d_in[16]; const float* rin_b = (const float*)d_in[17];
  const float* rln_g = (const float*)d_in[18]; const float* rln_b = (const float*)d_in[19];
  const float* rm1_w = (const float*)d_in[20]; const float* rm1_b = (const float*)d_in[21];
  const float* rm2_w = (const float*)d_in[22]; const float* rm2_b = (const float*)d_in[23];
  const float* rca_w = (const float*)d_in[24]; const float* rca_b = (const float*)d_in[25];
  float* out = (float*)d_out;

  // workspace layout (elements)
  const size_t BIG = (size_t)BB * CC * HWH;       // 15,728,640 f32
  float* vbuf = (float*)d_ws;                     // v
  float* xq   = vbuf + BIG;                       // x.Wq (no bias) ; later dw1 temp
  float* xk   = vbuf + 2 * BIG;                   // x.Wk (no bias) ; later pre-out
  float* vs   = vbuf + 3 * BIG;                   // group conv output
  float* ao   = vbuf + 4 * BIG;                   // attention output
  _Float16* wpv = (_Float16*)(vbuf + 5 * BIG);
  _Float16* wpq = wpv + 4096;
  _Float16* wpk = wpq + 4096;
  _Float16* wpo = wpk + 4096;
  float* maskb = (float*)(wpo + 4096);            // B*NW
  float* psum  = maskb + BB * NWIN;               // B*NW*16
  float* cab   = psum + (size_t)BB * NWIN * 16;   // B*CC

  const int gemm_blocks = (BB * HWH) / 128;       // 2048
  const int elem_blocks = (BB * CC * HWH) / 256;  // 61440

  // weight conversion f32 -> padded f16
  k_cvt_w<<<16, 256, 0, stream>>>(pv_w, wpv);
  k_cvt_w<<<16, 256, 0, stream>>>(pq_w, wpq);
  k_cvt_w<<<16, 256, 0, stream>>>(pk_w, wpk);
  k_cvt_w<<<16, 256, 0, stream>>>(po_w, wpo);

  // pointwise GEMMs (WMMA)
  k_gemm_c1x1<<<gemm_blocks, 256, 0, stream>>>(x, wpv, pv_b, vbuf, 1);
  k_gemm_c1x1<<<gemm_blocks, 256, 0, stream>>>(x, wpq, pq_b, xq, 0);
  k_gemm_c1x1<<<gemm_blocks, 256, 0, stream>>>(x, wpk, pk_b, xk, 0);

  // routing predictor + channel attention
  k_predictor<<<BB * NWIN, 64, 0, stream>>>(vbuf, cg, rin_w, rin_b, rln_g, rln_b,
                                            rm1_w, rm1_b, rm2_w, rm2_b, maskb, psum);
  k_ca<<<BB, 64, 0, stream>>>(psum, rca_w, rca_b, cab);

  // LSConv stand-in: 3x3 group conv
  k_gconv<<<elem_blocks, 256, 0, stream>>>(vbuf, ls_w, ls_b, vs);

  // window attention (WMMA x2 GEMMs per window) fused with (1-mask)*vs
  k_attn<<<BB * NWIN, 128, 0, stream>>>(xq, xk, vbuf, vs, maskb, pq_b, pk_b, ao);

  // conv-spatial refine: dw3x3(d1) -> dw3x3(d2) + gelu*ca + residual
  k_dwconv1<<<elem_blocks, 256, 0, stream>>>(ao, cs1_w, cs1_b, xq);      // xq = cs1
  k_dw2_gelu<<<elem_blocks, 256, 0, stream>>>(xq, cs2_w, cs2_b, ao, cab, xk); // xk = pre

  // final projection (WMMA)
  k_gemm_c1x1<<<gemm_blocks, 256, 0, stream>>>(xk, wpo, po_b, out, 1);
}
